// TFAttention_32530082300308
// MI455X (gfx1250) — matrix operand order repaired and checked by reading, not compiled, not run on hardware
//
#include <hip/hip_runtime.h>

typedef __attribute__((ext_vector_type(16))) _Float16 v16h;
typedef __attribute__((ext_vector_type(8)))  _Float16 v8h;
typedef __attribute__((ext_vector_type(8)))  float    v8f;
typedef __attribute__((ext_vector_type(4)))  unsigned u32x4;
typedef __attribute__((ext_vector_type(8)))  int      i32x8;
typedef __attribute__((ext_vector_type(4)))  int      i32x4;

static constexpr int Bc = 2, Sc = 2048, Dc = 1024, Hc = 16, DHc = 64;

#define WMMA_F16(a, b, c) \
  __builtin_amdgcn_wmma_f32_16x16x32_f16(false, (a), false, (b), (short)0, (c), false, false)

static __device__ inline v8f zero8() {
  v8f z;
#pragma unroll
  for (int i = 0; i < 8; ++i) z[i] = 0.0f;
  return z;
}

// DPP ROW_XMASK cross-lane max within 16-lane rows (wave32 halves disjoint).
template <int XMASK>
static __device__ inline float dpp_xor_max(float v) {
  int s = __builtin_bit_cast(int, v);
  int t = __builtin_amdgcn_update_dpp(s, s, 0x160 | XMASK, 0xf, 0xf, true);
  return fmaxf(v, __builtin_bit_cast(float, t));
}
static __device__ inline float rowmax16(float v) {
  v = dpp_xor_max<1>(v);
  v = dpp_xor_max<2>(v);
  v = dpp_xor_max<4>(v);
  v = dpp_xor_max<8>(v);
  return v;
}

// Generic -> LDS address-space cast to get the raw LDS byte offset for TDM.
static __device__ inline unsigned lds_addr_of(const void* p) {
  return (unsigned)(unsigned long long)(const __attribute__((address_space(3))) char*)p;
}

// TDM: async-load a 2D f16 tile [rows x 64] (row stride 3072 f16) into LDS,
// padding 4 DWORDs after every 32 DWORDs (=> 72-halfword LDS row stride).
static __device__ inline void tdm_load_tile_k(const _Float16* gsrc, unsigned lds_byte_off,
                                              int rows) {
  unsigned long long ga = (unsigned long long)gsrc;
  u32x4 g0;
  g0[0] = 1u;                                          // count=1, user mode
  g0[1] = lds_byte_off;                                // lds_addr
  g0[2] = (unsigned)(ga & 0xffffffffu);                // global_addr[31:0]
  g0[3] = (unsigned)((ga >> 32) & 0x01ffffffu) | (2u << 30);  // addr[56:32] | type=2
  i32x8 g1;
  g1[0] = (1 << 16)        // data_size = 2 bytes
        | (1 << 20)        // pad_enable
        | (4 << 22)        // pad_interval: 32 DWORDs
        | (3 << 25);       // pad_amount: 4 DWORDs
  g1[1] = (DHc & 0xffff) << 16;        // tensor_dim0[15:0] at bits[63:48]
  g1[2] = (Sc & 0xffff) << 16;         // tensor_dim1[15:0] at bits[95:80]
  g1[3] = (DHc & 0xffff) << 16;        // tile_dim0 at bits[127:112]
  g1[4] = rows & 0xffff;               // tile_dim1 at bits[143:128]
  g1[5] = 3 * Dc;                      // tensor_dim0_stride[31:0]
  g1[6] = 0;
  g1[7] = 0;
  i32x4 z4 = {0, 0, 0, 0};
  i32x8 z8 = {0, 0, 0, 0, 0, 0, 0, 0};
  __builtin_amdgcn_tensor_load_to_lds(g0, g1, z4, z4, z8, 0);
}

// A-fragment (16x32 f16, M x K): lane l -> row (l&15):
//   halves 0..7  = K = ((l>>4)*8) + 0..7 ; halves 8..15 = +16
static __device__ inline v16h lds_a_frag(const _Float16* base, int stride, int k0) {
  const int lane = threadIdx.x & 31;
  const _Float16* p = base + (lane & 15) * stride + k0 + ((lane >> 4) << 3);
  v8h lo = *(const v8h*)(p);
  v8h hi = *(const v8h*)(p + 16);
  v16h r;
#pragma unroll
  for (int i = 0; i < 8; ++i) { r[i] = lo[i]; r[i + 8] = hi[i]; }
  return r;
}

// B-fragment (32x16 f16, K x N), stored [n][k]: lane l -> col (l&15),
// halves 0..7 = K = ((l>>4)*8) + 0..7 ; halves 8..15 = +16
static __device__ inline v16h lds_b_frag(const _Float16* base, int stride, int k0) {
  const int lane = threadIdx.x & 31;
  const _Float16* p = base + (lane & 15) * stride + k0 + ((lane >> 4) << 3);
  v8h lo = *(const v8h*)(p);
  v8h hi = *(const v8h*)(p + 16);
  v16h r;
#pragma unroll
  for (int i = 0; i < 8; ++i) { r[i] = lo[i]; r[i + 8] = hi[i]; }
  return r;
}

// ---------------------------------------------------------------------------
// Tiled GEMM: C[M,N] = A[M,K] * B[K,N] + bias[N]; 256 thr, 128x128 tile, BK=32
// ---------------------------------------------------------------------------
template <typename TA, bool OUT_F16>
__global__ __launch_bounds__(256) void gemm128x128(
    const TA* __restrict__ A, const float* __restrict__ Bm,
    const float* __restrict__ bias, void* __restrict__ Cout,
    int M, int N, int K) {
  constexpr int BM = 128, BN = 128, BK = 32;
  constexpr int AST = BK + 8;
  constexpr int BST = BK + 8;
  __shared__ __align__(16) _Float16 As[BM * AST];
  __shared__ __align__(16) _Float16 Bs[BN * BST];  // transposed: [n][k]

  const int tid = threadIdx.x;
  const int lane = tid & 31;
  const int wave = tid >> 5;
  const int wm = wave >> 2;
  const int wn = wave & 3;
  const int row0 = blockIdx.y * BM;
  const int col0 = blockIdx.x * BN;

  v8f acc[4][2];
#pragma unroll
  for (int i = 0; i < 4; ++i)
#pragma unroll
    for (int j = 0; j < 2; ++j) acc[i][j] = zero8();

  for (int kk = 0; kk < K; kk += BK) {
    // Stage A tile (convert to f16): 128 rows x 32 k
    {
      const int r = tid >> 1;
      const int ks = (tid & 1) * 16;
      const TA* src = A + (size_t)(row0 + r) * K + kk + ks;
      _Float16* dst = As + r * AST + ks;
#pragma unroll
      for (int i = 0; i < 16; ++i) dst[i] = (_Float16)src[i];
      // Speculative GL2 prefetch of next k-chunk's A line
      if (kk + BK < K) __builtin_prefetch(src + BK, 0, 1);
    }
    // Stage B tile transposed with packed-pair b32 stores
    {
      const int k2 = (tid >> 4) * 2;       // 0..30
      const int ns = (tid & 15) * 8;       // 0..120
      const float* s0 = Bm + (size_t)(kk + k2) * N + col0 + ns;
      const float* s1 = s0 + N;
#pragma unroll
      for (int i = 0; i < 8; ++i) {
        union { _Float16 h[2]; unsigned u; } pk;
        pk.h[0] = (_Float16)s0[i];
        pk.h[1] = (_Float16)s1[i];
        *(unsigned*)&Bs[(ns + i) * BST + k2] = pk.u;
      }
      // Speculative GL2 prefetch of next k-chunk's B lines
      if (kk + BK < K) __builtin_prefetch(s0 + (size_t)BK * N, 0, 1);
    }
    __syncthreads();

    v16h af[4], bf[2];
#pragma unroll
    for (int ms = 0; ms < 4; ++ms)
      af[ms] = lds_a_frag(As + (wm * 64 + ms * 16) * AST, AST, 0);
#pragma unroll
    for (int ns = 0; ns < 2; ++ns)
      bf[ns] = lds_b_frag(Bs + (wn * 32 + ns * 16) * BST, BST, 0);
#pragma unroll
    for (int ms = 0; ms < 4; ++ms)
#pragma unroll
      for (int ns = 0; ns < 2; ++ns)
        acc[ms][ns] = WMMA_F16(af[ms], bf[ns], acc[ms][ns]);
    __syncthreads();
  }

#pragma unroll
  for (int ms = 0; ms < 4; ++ms) {
#pragma unroll
    for (int ns = 0; ns < 2; ++ns) {
      const int col = col0 + wn * 32 + ns * 16 + (lane & 15);
      const float bv = bias[col];
      const int rbase = row0 + wm * 64 + ms * 16 + ((lane >> 4) << 3);
#pragma unroll
      for (int r = 0; r < 8; ++r) {
        const float v = acc[ms][ns][r] + bv;
        const size_t idx = (size_t)(rbase + r) * N + col;
        if (OUT_F16)
          ((_Float16*)Cout)[idx] = (_Float16)v;
        else
          ((float*)Cout)[idx] = v;
      }
    }
  }
}

// ---------------------------------------------------------------------------
// Flash attention, causal (masked = exactly -10000 like the reference).
// 128 threads (4 waves) per 64 query rows of one (b,h).
// K tiles double-buffered via the Tensor Data Mover (wave 0 issues, prefetch
// of tile kt+1 overlaps compute of tile kt; s_wait_tensorcnt 1 keeps the
// current tile resident while the prefetch is in flight).
// ---------------------------------------------------------------------------
__global__ __launch_bounds__(128) void flash_attn(
    const _Float16* __restrict__ qkv, _Float16* __restrict__ out) {
  constexpr int QT = 64, KT = 32, DH = DHc;
  constexpr int QS = DH + 8;  // 72 f16 stride (matches TDM pad 4dw/32dw)
  constexpr int KS = DH + 8;
  constexpr int VS = KT + 8;
  constexpr int PS = KT + 8;
  __shared__ __align__(16) _Float16 Qs[QT * QS];
  __shared__ __align__(16) _Float16 Ks[2 * KT * KS];  // double-buffered TDM dest
  __shared__ __align__(16) _Float16 Vt[DH * VS];      // transposed: [d][k]
  __shared__ __align__(16) _Float16 Pw[4 * 16 * PS];  // per-wave P patch

  const int tid = threadIdx.x;
  const int lane = tid & 31;
  const int wave = tid >> 5;
  const int qt = blockIdx.x, h = blockIdx.y, b = blockIdx.z;
  const int qbase = qt * QT;
  const size_t rowStride = (size_t)3 * Dc;
  const _Float16* qp = qkv + (size_t)b * Sc * rowStride + h * DHc;
  const _Float16* kp = qp + Dc;
  const _Float16* vp = qp + 2 * Dc;

  const int nkt = (qbase + QT) / KT;

  // Kick off the first K-tile DMA before anything else (maximum overlap).
  if (wave == 0) {
    tdm_load_tile_k(kp, lds_addr_of(Ks), KT);
  }

  // Stage Q tile: 64 rows x 64 d
  {
    const int r = tid >> 1;
    const int d0 = (tid & 1) * 32;
    const _Float16* src = qp + (size_t)(qbase + r) * rowStride + d0;
    _Float16* dst = Qs + r * QS + d0;
#pragma unroll
    for (int i = 0; i < 32; ++i) dst[i] = src[i];
  }
  __syncthreads();

  v16h aq[2];
  aq[0] = lds_a_frag(Qs + wave * 16 * QS, QS, 0);
  aq[1] = lds_a_frag(Qs + wave * 16 * QS, QS, 32);

  v16h bones;  // ones B-operand for WMMA-based row sums
#pragma unroll
  for (int i = 0; i < 16; ++i) bones[i] = (_Float16)1.0f;

  float m_run[8], lsum[8];
  v8f o[4];
#pragma unroll
  for (int r = 0; r < 8; ++r) { m_run[r] = -3.0e38f; lsum[r] = 0.0f; }
#pragma unroll
  for (int j = 0; j < 4; ++j) o[j] = zero8();

  const int rowoff = qbase + wave * 16 + ((lane >> 4) << 3);

  for (int kt = 0; kt < nkt; ++kt) {
    // Prefetch next K tile into the other buffer; ensure current tile landed.
    if (wave == 0) {
      if (kt + 1 < nkt) {
        tdm_load_tile_k(kp + (size_t)((kt + 1) * KT) * rowStride,
                        lds_addr_of(Ks + ((kt + 1) & 1) * KT * KS), KT);
        __builtin_amdgcn_s_wait_tensorcnt(1);  // in-order: tile kt is resident
      } else {
        __builtin_amdgcn_s_wait_tensorcnt(0);
      }
    }
    const _Float16* ks = Ks + (kt & 1) * KT * KS;

    // V tile transposed, packed-pair b32 stores: 2 rows x 8 d per thread
    {
      const int r2 = (tid & 15) * 2;     // 0..30
      const int d0 = (tid >> 4) * 8;     // 0..56
      const _Float16* s0 = vp + (size_t)(kt * KT + r2) * rowStride + d0;
      const _Float16* s1 = s0 + rowStride;
#pragma unroll
      for (int i = 0; i < 8; ++i) {
        union { _Float16 h[2]; unsigned u; } pk;
        pk.h[0] = s0[i];
        pk.h[1] = s1[i];
        *(unsigned*)&Vt[(d0 + i) * VS + r2] = pk.u;
      }
    }
    __syncthreads();

    // Scores: S[16x32] = Q @ K^T
    v8f s[2] = {zero8(), zero8()};
#pragma unroll
    for (int kc = 0; kc < 2; ++kc) {
#pragma unroll
      for (int nf = 0; nf < 2; ++nf) {
        v16h bb = lds_b_frag(ks + nf * 16 * KS, KS, kc * 32);
        s[nf] = WMMA_F16(aq[kc], bb, s[nf]);
      }
    }

    // Scale + causal mask
#pragma unroll
    for (int nf = 0; nf < 2; ++nf) {
      const int col = kt * KT + nf * 16 + (lane & 15);
#pragma unroll
      for (int r = 0; r < 8; ++r) {
        const float sv = s[nf][r] * 0.125f;
        s[nf][r] = (col <= rowoff + r) ? sv : -10000.0f;
      }
    }

    // Online softmax: row max via DPP ROW_XMASK (VALU only)
    float alpha[8], mnew[8];
#pragma unroll
    for (int r = 0; r < 8; ++r) {
      float t = rowmax16(fmaxf(s[0][r], s[1][r]));
      const float mn = fmaxf(m_run[r], t);
      alpha[r] = __expf(m_run[r] - mn);
      m_run[r] = mn;
      mnew[r] = mn;
    }

    // P = exp(S - m) -> per-wave LDS patch (C-layout -> A-layout bounce)
    _Float16* pw = Pw + wave * 16 * PS;
#pragma unroll
    for (int nf = 0; nf < 2; ++nf) {
#pragma unroll
      for (int r = 0; r < 8; ++r) {
        const float p = __expf(s[nf][r] - mnew[r]);
        pw[(((lane >> 4) << 3) + r) * PS + nf * 16 + (lane & 15)] = (_Float16)p;
      }
    }
#pragma unroll
    for (int j = 0; j < 4; ++j)
#pragma unroll
      for (int r = 0; r < 8; ++r) o[j][r] *= alpha[r];

    __syncthreads();  // publish Pw, order before PV reads

    v16h pa = lds_a_frag(pw, PS, 0);
    // Row sums via WMMA with ones B-operand: slot r == rowsum(row r+8*(l>>4))
    v8f rs = WMMA_F16(pa, bones, zero8());
#pragma unroll
    for (int r = 0; r < 8; ++r) lsum[r] = lsum[r] * alpha[r] + rs[r];

    // O += P @ V
#pragma unroll
    for (int j = 0; j < 4; ++j) {
      v16h vb = lds_b_frag(Vt + j * 16 * VS, VS, 0);
      o[j] = WMMA_F16(pa, vb, o[j]);
    }
    __syncthreads();  // protect Vt / retire Ks[kt&1] readers before reuse
  }

  // Normalize + store merged-heads f16: out[b, q, h*64 + d]
#pragma unroll
  for (int r = 0; r < 8; ++r) {
    const float inv = 1.0f / lsum[r];
    const int q = rowoff + r;
#pragma unroll
    for (int j = 0; j < 4; ++j) {
      const int d = j * 16 + (lane & 15);
      out[((size_t)b * Sc + q) * Dc + h * DHc + d] = (_Float16)(o[j][r] * inv);
    }
  }
}

extern "C" void kernel_launch(void* const* d_in, const int* in_sizes, int n_in,
                              void* d_out, int out_size, void* d_ws, size_t ws_size,
                              hipStream_t stream) {
  const float* x      = (const float*)d_in[0];
  const float* w_attn = (const float*)d_in[1];
  const float* b_attn = (const float*)d_in[2];
  const float* w_proj = (const float*)d_in[3];
  const float* b_proj = (const float*)d_in[4];

  _Float16* qkv  = (_Float16*)d_ws;                 // [B*S, 3D] f16
  _Float16* attn = qkv + (size_t)Bc * Sc * 3 * Dc;  // [B*S, D]  f16

  {
    dim3 grid(3 * Dc / 128, (Bc * Sc) / 128);
    gemm128x128<float, true><<<grid, 256, 0, stream>>>(
        x, w_attn, b_attn, (void*)qkv, Bc * Sc, 3 * Dc, Dc);
  }
  {
    dim3 grid(Sc / 64, Hc, Bc);
    flash_attn<<<grid, 128, 0, stream>>>(qkv, attn);
  }
  {
    dim3 grid(Dc / 128, (Bc * Sc) / 128);
    gemm128x128<_Float16, false><<<grid, 256, 0, stream>>>(
        attn, w_proj, b_proj, d_out, Bc * Sc, Dc, Dc);
  }
}
